// CRF_55800215110144
// MI455X (gfx1250) — matrix operand order repaired and checked by reading, not compiled, not run on hardware
//
#include <hip/hip_runtime.h>
#include <math.h>

typedef __attribute__((ext_vector_type(16))) _Float16 v16h;
typedef __attribute__((ext_vector_type(8)))  float    v8f;
typedef unsigned int u32x4 __attribute__((ext_vector_type(4)));

#define NEGV  (-10000.0f)
#define Bsz   512
#define Ssz   1024
#define NLsz  49
#define Lsz   51          // NL + 2
#define LP    64          // padded state count (WMMA M/K)
#define TB    16          // batches per wave-tile (WMMA N)
#define PROW  72          // padded shPT row (halves): 144B stride, conflict-free

// ---------------------------------------------------------------------------
// Kernel 1: gold path score per batch  -> gold_ws[b]
// ---------------------------------------------------------------------------
__global__ __launch_bounds__(256)
void crf_gold_kernel(const float* __restrict__ logits,
                     const float* __restrict__ trans,
                     const int*   __restrict__ labels,
                     const int*   __restrict__ lens,
                     float*       __restrict__ gold_ws)
{
    __shared__ float red[256];
    const int b   = blockIdx.x;
    const int tid = threadIdx.x;
    const int len = lens[b];
    const int*   lab = labels + (size_t)b * Ssz;
    const float* lg  = logits + (size_t)b * Ssz * NLsz;

    float acc = 0.0f;
    for (int t = tid; t < len; t += 256) {
        int cur  = lab[t];
        int prev = (t == 0) ? (Lsz - 2) : lab[t - 1];   // start = L-2
        acc += lg[(size_t)t * NLsz + cur];
        acc += trans[cur * Lsz + prev];
    }
    if (tid == 0) acc += trans[(Lsz - 1) * Lsz + lab[len - 1]];  // end transition

    red[tid] = acc;
    __syncthreads();
    for (int s = 128; s > 0; s >>= 1) {
        if (tid < s) red[tid] += red[tid + s];
        __syncthreads();
    }
    if (tid == 0) gold_ws[b] = red[0];
}

// ---------------------------------------------------------------------------
// Kernel 2: scaled-forward scan, fully in exp domain.
// State per batch: normalized vector p (f16 in LDS) + log-scale C (VGPR).
// Per step: S = E @ p (8x v_wmma_f32_16x16x32_f16),
//           p' = (S/maxS) * exp(lg_t - maxLg),  C += maxLg + log(maxS).
// exp(lg) factors for step t+1 are computed during step t (pipelined).
// ---------------------------------------------------------------------------
__global__ __launch_bounds__(32)
void crf_scan_kernel(const float* __restrict__ logits,
                     const float* __restrict__ trans,
                     const int*   __restrict__ lens,
                     const float* __restrict__ gold_ws,
                     float*       __restrict__ out)
{
    __shared__ _Float16 shE[LP][LP];                      // exp(T), zero-padded
    __shared__ __align__(16) _Float16 shPT[TB][PROW];     // p, batch-major

    const int l     = threadIdx.x;            // 0..31
    const int b0    = blockIdx.x * TB;
    const int n     = l & 15;                 // WMMA column = batch-in-tile
    const int off   = (l < 16) ? 0 : 8;       // C/D row offset for this half-wave
    const int bbase = (l < 16) ? 0 : 8;       // B-matrix K offset for this half-wave
    const int myLen = lens[b0 + n];
    const float* rowPtr = logits + (size_t)(b0 + n) * Ssz * NLsz;

    // E = exp(T) once (f16; NEG rows/cols and padding -> exact 0)
    for (int e = l; e < LP * LP; e += 32) {
        int j = e >> 6, k = e & 63;
        float v = 0.0f;
        if (j < Lsz && k < Lsz) v = __expf(trans[j * Lsz + k]);
        shE[j][k] = (_Float16)v;
    }
    // p0 = one-hot at start state (alpha0 = NEG except start=0, C0 = 0)
    for (int e = l; e < TB * LP; e += 32) {
        int nn = e >> 6, k = e & 63;
        shPT[nn][k] = (k == Lsz - 2) ? (_Float16)1.0f : (_Float16)0.0f;
    }
    __syncthreads();

    // Constant A-fragments afr[kt][mt] (ISA 7.12.2 16-bit A layout)
    v16h afr[2][4];
    const int abase = (l < 16) ? 0 : 8;
    #pragma unroll
    for (int kt = 0; kt < 2; ++kt) {
        #pragma unroll
        for (int mt = 0; mt < 4; ++mt) {
            v16h a;
            #pragma unroll
            for (int i = 0; i < 16; ++i) {
                int v = i >> 1, h = i & 1;
                int kk = ((v < 4) ? (2 * v + h) : (16 + 2 * (v - 4) + h))
                         + abase + 32 * kt;
                a[i] = shE[(l & 15) + 16 * mt][kk];
            }
            afr[kt][mt] = a;
        }
    }

    union BU { u32x4 q[2]; v16h h; };
    union PK { _Float16 h[8]; u32x4 u; };

    // Prologue: exp-factors for step 0
    float eLg[4][8];
    float maxLg;
    float C = 0.0f;
    {
        float lgv[4][8], mx = -3.0e38f;
        #pragma unroll
        for (int mt = 0; mt < 4; ++mt)
            #pragma unroll
            for (int r = 0; r < 8; ++r) {
                int j = 16 * mt + off + r;
                float v = -3.0e38f;
                if (j < NLsz) v = rowPtr[j];
                lgv[mt][r] = v;
                mx = fmaxf(mx, v);
            }
        mx = fmaxf(mx, __shfl_xor(mx, 16, 32));
        maxLg = mx;
        #pragma unroll
        for (int mt = 0; mt < 4; ++mt)
            #pragma unroll
            for (int r = 0; r < 8; ++r) {
                int j = 16 * mt + off + r;
                eLg[mt][r] = (j < NLsz) ? __expf(lgv[mt][r] - mx) : 0.0f;
            }
    }

    for (int t = 0; t < Ssz; ++t) {
        // (1) Issue next step's logit loads first (hidden behind WMMA)
        float lgv[4][8];
        if (t + 1 < Ssz) {
            const float* p = rowPtr + (size_t)(t + 1) * NLsz;
            #pragma unroll
            for (int mt = 0; mt < 4; ++mt)
                #pragma unroll
                for (int r = 0; r < 8; ++r) {
                    int j = 16 * mt + off + r;
                    lgv[mt][r] = (j < NLsz) ? p[j] : -3.0e38f;
                }
        }

        // (2) S = E @ p : contiguous b128 B-fragment loads + 8 WMMA
        v8f acc[4] = { v8f{}, v8f{}, v8f{}, v8f{} };
        #pragma unroll
        for (int kt = 0; kt < 2; ++kt) {
            BU bu;
            bu.q[0] = *(const u32x4*)&shPT[n][bbase + 32 * kt];
            bu.q[1] = *(const u32x4*)&shPT[n][bbase + 32 * kt + 16];
            #pragma unroll
            for (int mt = 0; mt < 4; ++mt) {
                acc[mt] = __builtin_amdgcn_wmma_f32_16x16x32_f16(
                    false, afr[kt][mt], false, bu.h,
                    (short)0, acc[mt], false, false);
            }
        }

        // (3) column max of S (in-lane tree + cross-half shuffle), premul by eLg
        float q[4][8], pm[4];
        #pragma unroll
        for (int mt = 0; mt < 4; ++mt) {
            float m = 0.0f;                       // S >= 0
            #pragma unroll
            for (int r = 0; r < 8; ++r) {
                float s = acc[mt][r];
                q[mt][r] = s * eLg[mt][r];
                m = fmaxf(m, s);
            }
            pm[mt] = m;
        }
        float mx = fmaxf(fmaxf(pm[0], pm[1]), fmaxf(pm[2], pm[3]));
        mx = fmaxf(mx, __shfl_xor(mx, 16, 32));
        float rs = __builtin_amdgcn_rcpf(mx);

        // (4) write p' and update C; frozen once t >= len (reference mask)
        if (t < myLen) {
            #pragma unroll
            for (int mt = 0; mt < 4; ++mt) {
                PK pk;
                #pragma unroll
                for (int r = 0; r < 8; ++r)
                    pk.h[r] = (_Float16)(q[mt][r] * rs);
                *(u32x4*)&shPT[n][16 * mt + off] = pk.u;
            }
            C += maxLg + __logf(mx);
        }

        // (5) finish next step's exp factors (off the p'->WMMA critical path)
        if (t + 1 < Ssz) {
            float ml = -3.0e38f;
            #pragma unroll
            for (int mt = 0; mt < 4; ++mt)
                #pragma unroll
                for (int r = 0; r < 8; ++r) ml = fmaxf(ml, lgv[mt][r]);
            ml = fmaxf(ml, __shfl_xor(ml, 16, 32));
            #pragma unroll
            for (int mt = 0; mt < 4; ++mt)
                #pragma unroll
                for (int r = 0; r < 8; ++r) {
                    int j = 16 * mt + off + r;
                    eLg[mt][r] = (j < NLsz) ? __expf(lgv[mt][r] - ml) : 0.0f;
                }
            maxLg = ml;
        }
    }
    __syncthreads();

    // Epilogue: norm = C + log( sum_j p[j] * exp(T[end, j]) );  out = gold - norm
    if (l < TB) {
        float s = 0.0f;
        for (int j = 0; j < Lsz; ++j)
            s += (float)shPT[l][j] * __expf(trans[(Lsz - 1) * Lsz + j]);
        out[b0 + l] = gold_ws[b0 + l] - (C + __logf(s));
    }
}

// ---------------------------------------------------------------------------
extern "C" void kernel_launch(void* const* d_in, const int* in_sizes, int n_in,
                              void* d_out, int out_size, void* d_ws, size_t ws_size,
                              hipStream_t stream) {
    const float* logits = (const float*)d_in[0];   // (512,1024,49) f32
    const float* trans  = (const float*)d_in[1];   // (51,51) f32
    const int*   labels = (const int*)d_in[2];     // (512,1024) i32
    const int*   lens   = (const int*)d_in[3];     // (512,) i32
    float* out  = (float*)d_out;                   // (512,) f32
    float* gold = (float*)d_ws;                    // 512 f32 scratch

    crf_gold_kernel<<<Bsz, 256, 0, stream>>>(logits, trans, labels, lens, gold);
    crf_scan_kernel<<<Bsz / TB, 32, 0, stream>>>(logits, trans, lens, gold, out);
}
